// LSTMTFAutoencoder_9448928051697
// MI455X (gfx1250) — matrix operand order repaired and checked by reading, not compiled, not run on hardware
//
#include <hip/hip_runtime.h>
#include <hip/hip_bf16.h>

// ---------------------------------------------------------------------------
// LSTM teacher-forcing autoencoder for MI455X (gfx1250, wave32, WMMA bf16)
// B=128, T=1024, D=64, H=256, L=2
// Fused gate design: each wave owns the i/f/g/o tiles of a (16-row,16-col)
// hidden block; c-state and h-state live in VGPRs for all 2048 steps.
// Weight pointers are laundered per time step (asm volatile zero offset) so
// LICM cannot hoist the fragment loads out of the loop (and spill them),
// while the loads remain ordinary cached global_load_b128s.
// ---------------------------------------------------------------------------

typedef __attribute__((ext_vector_type(16))) __bf16 v16bf;
typedef __attribute__((ext_vector_type(8)))  __bf16 v8bf;
typedef __attribute__((ext_vector_type(8)))  float  v8f;

#define B_   128
#define T_   1024
#define D_   64
#define H_   256
#define G4   1024
#define NWG  16            // persistent grid (16 WGs x 8 wave32 = 128 waves)
#define TPB  256
#define ASTR 264           // bf16 LDS row stride (16B-aligned rows, staggered banks)

// opaque zero: compiler must assume it changes every time this executes
__device__ __forceinline__ size_t opaque_zero() {
  size_t z = 0;
  asm volatile("" : "+s"(z));
  return z;
}

// ---------------------------------------------------------------------------
// grid-wide sense-reversing barrier
// ---------------------------------------------------------------------------
__device__ __forceinline__ void gbar(unsigned* cnt, unsigned* gen) {
  __syncthreads();
  if (threadIdx.x == 0) {
    __threadfence();
    unsigned g = __hip_atomic_load(gen, __ATOMIC_ACQUIRE, __HIP_MEMORY_SCOPE_AGENT);
    unsigned a = __hip_atomic_fetch_add(cnt, 1u, __ATOMIC_ACQ_REL, __HIP_MEMORY_SCOPE_AGENT);
    if (a == NWG - 1u) {
      __hip_atomic_store(cnt, 0u, __ATOMIC_RELAXED, __HIP_MEMORY_SCOPE_AGENT);
      __hip_atomic_fetch_add(gen, 1u, __ATOMIC_RELEASE, __HIP_MEMORY_SCOPE_AGENT);
    } else {
      while (__hip_atomic_load(gen, __ATOMIC_ACQUIRE, __HIP_MEMORY_SCOPE_AGENT) == g)
        __builtin_amdgcn_s_sleep(2);
    }
    __threadfence();
  }
  __syncthreads();
}

// ---------------------------------------------------------------------------
// fragment helpers
// ---------------------------------------------------------------------------
__device__ __forceinline__ v8f wmma_bf16(v16bf a, v16bf b, v8f c) {
  return __builtin_amdgcn_wmma_f32_16x16x32_bf16(false, a, false, b, (short)0, c,
                                                 false, false);
}

// A-fragment from a bf16 LDS tile: lane holds row M=lane&15; element j maps to
// K = ks + (j&7) + ((j>>3)<<4) + (lane>=16 ? 8 : 0)  -> two contiguous 16B loads.
__device__ __forceinline__ v16bf afrag_lds(const __bf16 (*act)[ASTR], int ks, unsigned lane) {
  const int hi8 = (lane & 16u) ? 8 : 0;
  const __bf16* row = act[lane & 15u];
  v8bf lo = *(const v8bf*)&row[ks + hi8];
  v8bf hi = *(const v8bf*)&row[ks + 16 + hi8];
  return __builtin_shufflevector(lo, hi, 0,1,2,3,4,5,6,7,8,9,10,11,12,13,14,15);
}

// A-fragment gathered straight from a global bf16 matrix (row stride = C)
__device__ __forceinline__ v16bf afrag_glb(const __bf16* __restrict__ m, int C,
                                           int row0, int ks, unsigned lane) {
  const int hi8 = (lane & 16u) ? 8 : 0;
  const __bf16* row = m + (size_t)(row0 + (int)(lane & 15u)) * C;
  v8bf lo = *(const v8bf*)&row[ks + hi8];
  v8bf hi = *(const v8bf*)&row[ks + 16 + hi8];
  return __builtin_shufflevector(lo, hi, 0,1,2,3,4,5,6,7,8,9,10,11,12,13,14,15);
}

// B-fragment from row-major bf16 weights: lane = column N (weight row n0+lane&15),
// K = ks + (j&7) + ((j>>3)<<4) + (lane>=16 ? 8 : 0) -> two aligned 16-byte cached loads.
__device__ __forceinline__ v16bf bfrag(const __bf16* __restrict__ W, int C,
                                       int n0, int ks, unsigned lane) {
  const __bf16* p = W + (size_t)(n0 + (int)(lane & 15u)) * C + ks +
                    ((lane & 16u) ? 8 : 0);
  v8bf lo = *(const v8bf*)p;
  v8bf hi = *(const v8bf*)(p + 16);
  return __builtin_shufflevector(lo, hi, 0,1,2,3,4,5,6,7,8,9,10,11,12,13,14,15);
}

__device__ __forceinline__ float sigm_(float v) { return 1.f / (1.f + __expf(-v)); }
__device__ __forceinline__ float tanh_(float v) { return 1.f - 2.f / (1.f + __expf(2.f * v)); }

// ---------------------------------------------------------------------------
// one LSTM layer step for this wave's (mt, ntH) block, gates fused in-register
// ---------------------------------------------------------------------------
__device__ __forceinline__ void lstm_layer(
    const __bf16* __restrict__ Wih, int kin, const __bf16* __restrict__ Whh,
    const __bf16 (*acta)[ASTR], const __bf16 (*acth)[ASTR],
    const float* __restrict__ bias,
    v8f& ctile, v8f& htile,
    __bf16* __restrict__ hout, float* __restrict__ enc_out,
    const int* lenr, int t, int mt, int ntH, unsigned lane, bool enc) {
  v8f zi = {}, zf = {}, zg = {}, zo = {};
  const int nb = ntH * 16;
#pragma unroll
  for (int ks = 0; ks < kin; ks += 32) {
    v16bf bi_ = bfrag(Wih, kin, nb,          ks, lane);
    v16bf bf_ = bfrag(Wih, kin, nb + H_,     ks, lane);
    v16bf bg_ = bfrag(Wih, kin, nb + 2 * H_, ks, lane);
    v16bf bo_ = bfrag(Wih, kin, nb + 3 * H_, ks, lane);
    v16bf a = afrag_lds(acta, ks, lane);
    zi = wmma_bf16(a, bi_, zi);
    zf = wmma_bf16(a, bf_, zf);
    zg = wmma_bf16(a, bg_, zg);
    zo = wmma_bf16(a, bo_, zo);
  }
#pragma unroll
  for (int ks = 0; ks < H_; ks += 32) {
    v16bf bi_ = bfrag(Whh, H_, nb,          ks, lane);
    v16bf bf_ = bfrag(Whh, H_, nb + H_,     ks, lane);
    v16bf bg_ = bfrag(Whh, H_, nb + 2 * H_, ks, lane);
    v16bf bo_ = bfrag(Whh, H_, nb + 3 * H_, ks, lane);
    v16bf a = afrag_lds(acth, ks, lane);
    zi = wmma_bf16(a, bi_, zi);
    zf = wmma_bf16(a, bf_, zf);
    zg = wmma_bf16(a, bg_, zg);
    zo = wmma_bf16(a, bo_, zo);
  }
  const int col = nb + (int)(lane & 15u);
  const float bi = bias[col], bff = bias[col + H_];
  const float bg = bias[col + 2 * H_], bo = bias[col + 3 * H_];
  const int m0 = mt * 16 + ((lane & 16u) ? 8 : 0);
#pragma unroll
  for (int r = 0; r < 8; ++r) {
    float cn = sigm_(zf[r] + bff) * ctile[r] + sigm_(zi[r] + bi) * tanh_(zg[r] + bg);
    float hn = sigm_(zo[r] + bo) * tanh_(cn);
    if (enc) {
      bool valid = t < lenr[r];
      hn = valid ? hn : htile[r];
      cn = valid ? cn : ctile[r];
      if (enc_out) enc_out[((size_t)(m0 + r) * T_ + t) * H_ + col] = valid ? hn : 0.f;
    }
    ctile[r] = cn;
    htile[r] = hn;
    hout[(size_t)(m0 + r) * H_ + col] = (__bf16)hn;
  }
}

// ---------------------------------------------------------------------------
// staging helpers: 16 batch rows (rowlo..rowlo+15) into LDS
// ---------------------------------------------------------------------------
__device__ __forceinline__ void stage_bf(__bf16 (*dst)[ASTR], const __bf16* __restrict__ src,
                                         int cols, int rowlo) {
  const int chunks = 16 * cols / 8;  // 16B chunks
  for (int idx = threadIdx.x; idx < chunks; idx += TPB) {
    int per = cols / 8;
    int r = idx / per, c8 = idx - r * per;
    *(v8bf*)&dst[r][c8 * 8] = *(const v8bf*)(src + (size_t)(rowlo + r) * cols + c8 * 8);
  }
}

__device__ __forceinline__ void stage_cvt(__bf16 (*dst)[ASTR], const float* __restrict__ src,
                                          int cols, int srcStride, int rowlo) {
  for (int idx = threadIdx.x; idx < 16 * cols; idx += TPB) {
    int r = idx / cols, c = idx - r * cols;
    dst[r][c] = (__bf16)src[(size_t)(rowlo + r) * srcStride + c];
  }
}

// ---------------------------------------------------------------------------
// projection: out = h1 @ outW^T + out_b; writes outputs (f32) + next decin (bf16)
// ---------------------------------------------------------------------------
__device__ __forceinline__ void proj_phase(const __bf16* __restrict__ wOut,
                                           const __bf16* __restrict__ h1,
                                           const float* __restrict__ out_b,
                                           const float* __restrict__ x,
                                           const unsigned char* __restrict__ tf_mask,
                                           float* __restrict__ outp,
                                           __bf16* __restrict__ decin,
                                           int t, unsigned wave_gid, unsigned lane) {
  const int mt = (int)(wave_gid >> 2);  // 0..7
  const int nt = (int)(wave_gid & 3u);  // 0..3
  v8f acc = {};
#pragma unroll
  for (int ks = 0; ks < H_; ks += 32) {
    v16bf b = bfrag(wOut, H_, nt * 16, ks, lane);
    v16bf a = afrag_glb(h1, H_, mt * 16, ks, lane);
    acc = wmma_bf16(a, b, acc);
  }
  const int col = nt * 16 + (int)(lane & 15u);
  const float bb = out_b[col];
  const bool tf = tf_mask[t] != 0;
  const int m0 = mt * 16 + ((lane & 16u) ? 8 : 0);
#pragma unroll
  for (int r = 0; r < 8; ++r) {
    int row = m0 + r;
    float o = acc[r] + bb;
    size_t oi = ((size_t)row * T_ + t) * D_ + col;
    outp[oi] = o;
    decin[(size_t)row * D_ + col] = tf ? (__bf16)x[oi] : (__bf16)o;
  }
}

// ---------------------------------------------------------------------------
// persistent recurrence kernel
// ---------------------------------------------------------------------------
__global__ __launch_bounds__(TPB) void lstm_persistent(
    const float* __restrict__ x, const int* __restrict__ lengths,
    const unsigned char* __restrict__ tf_mask,
    const float* __restrict__ eb0, const float* __restrict__ eb1,
    const float* __restrict__ db0, const float* __restrict__ db1,
    const float* __restrict__ outb,
    const __bf16* wEih0, const __bf16* wEhh0,
    const __bf16* wEih1, const __bf16* wEhh1,
    const __bf16* wDih0, const __bf16* wDhh0,
    const __bf16* wDih1, const __bf16* wDhh1,
    const __bf16* wOut,
    __bf16* h0g, __bf16* h1g, __bf16* decin,
    unsigned* bar_cnt, unsigned* bar_gen,
    float* outp, float* enc_out) {
  __shared__ __bf16 acta[16][ASTR];
  __shared__ __bf16 acth[16][ASTR];

  const unsigned tid = threadIdx.x, lane = tid & 31u, wv = tid >> 5, wg = blockIdx.x;
  const int mt = (int)(wg >> 1);                 // 0..7
  const int ntH = (int)((wg & 1u) * 8u + wv);    // 0..15
  const int rowlo = mt * 16;
  const unsigned wave_gid = wg * 8u + wv;

  int lenr[8];
  {
    const int m0 = mt * 16 + ((lane & 16u) ? 8 : 0);
#pragma unroll
    for (int r = 0; r < 8; ++r) lenr[r] = lengths[m0 + r];
  }

  v8f c0t = {}, h0t = {}, c1t = {}, h1t = {};
  int p = 0;

  // ----------------- encoder -----------------
  for (int t = 0; t < T_; ++t) {
    const size_t lz = opaque_zero();  // defeat LICM of weight loads, keep them cached
    stage_cvt(acta, x + (size_t)t * D_, D_, T_ * D_, rowlo);
    stage_bf(acth, h0g + (size_t)p * B_ * H_, H_, rowlo);
    __syncthreads();
    lstm_layer(wEih0 + lz, D_, wEhh0 + lz, acta, acth, eb0, c0t, h0t,
               h0g + (size_t)(p ^ 1) * B_ * H_, nullptr, lenr, t, mt, ntH, lane, true);
    gbar(bar_cnt, bar_gen);

    stage_bf(acta, h0g + (size_t)(p ^ 1) * B_ * H_, H_, rowlo);
    stage_bf(acth, h1g + (size_t)p * B_ * H_, H_, rowlo);
    __syncthreads();
    lstm_layer(wEih1 + lz, H_, wEhh1 + lz, acta, acth, eb1, c1t, h1t,
               h1g + (size_t)(p ^ 1) * B_ * H_, enc_out, lenr, t, mt, ntH, lane, true);
    gbar(bar_cnt, bar_gen);
    p ^= 1;
  }

  // ----------------- decoder -----------------
  for (int t = 0; t < T_; ++t) {
    const size_t lz = opaque_zero();
    stage_bf(acta, decin, D_, rowlo);
    stage_bf(acth, h0g + (size_t)p * B_ * H_, H_, rowlo);
    __syncthreads();
    lstm_layer(wDih0 + lz, D_, wDhh0 + lz, acta, acth, db0, c0t, h0t,
               h0g + (size_t)(p ^ 1) * B_ * H_, nullptr, nullptr, t, mt, ntH, lane, false);
    gbar(bar_cnt, bar_gen);

    stage_bf(acta, h0g + (size_t)(p ^ 1) * B_ * H_, H_, rowlo);
    stage_bf(acth, h1g + (size_t)p * B_ * H_, H_, rowlo);
    __syncthreads();
    lstm_layer(wDih1 + lz, H_, wDhh1 + lz, acta, acth, db1, c1t, h1t,
               h1g + (size_t)(p ^ 1) * B_ * H_, nullptr, nullptr, t, mt, ntH, lane, false);
    gbar(bar_cnt, bar_gen);

    if (wave_gid < 32u)  // wave-uniform: EXEC all-ones inside for WMMA
      proj_phase(wOut + lz, h1g + (size_t)(p ^ 1) * B_ * H_, outb, x, tf_mask,
                 outp, decin, t, wave_gid, lane);
    gbar(bar_cnt, bar_gen);
    p ^= 1;
  }
}

// ---------------------------------------------------------------------------
// setup kernels
// ---------------------------------------------------------------------------
__global__ void cvt_kernel(const float* __restrict__ src, __bf16* __restrict__ dst, int n) {
  for (int i = blockIdx.x * blockDim.x + threadIdx.x; i < n; i += gridDim.x * blockDim.x)
    dst[i] = (__bf16)src[i];
}

__global__ void init_kernel(__bf16* h0g, __bf16* h1g, __bf16* decin,
                            const float* __restrict__ x, unsigned* bar) {
  int idx = blockIdx.x * blockDim.x + threadIdx.x;
  if (idx < B_ * H_) {
    h0g[idx] = (__bf16)0.f; h0g[B_ * H_ + idx] = (__bf16)0.f;
    h1g[idx] = (__bf16)0.f; h1g[B_ * H_ + idx] = (__bf16)0.f;
  }
  if (idx < B_ * D_) {
    int b = idx / D_, d = idx - b * D_;
    decin[idx] = (__bf16)x[((size_t)b * T_ + (T_ - 1)) * D_ + d];
  }
  if (idx < 2) bar[idx] = 0u;
}

// ---------------------------------------------------------------------------
// host launch
// ---------------------------------------------------------------------------
extern "C" void kernel_launch(void* const* d_in, const int* in_sizes, int n_in,
                              void* d_out, int out_size, void* d_ws, size_t ws_size,
                              hipStream_t stream) {
  (void)in_sizes; (void)n_in; (void)out_size; (void)ws_size;
  const float* x       = (const float*)d_in[0];
  const int* lengths   = (const int*)d_in[1];
  const unsigned char* tf = (const unsigned char*)d_in[2];
  const float* eWih0 = (const float*)d_in[3];
  const float* eWhh0 = (const float*)d_in[4];
  const float* eb0   = (const float*)d_in[5];
  const float* eWih1 = (const float*)d_in[6];
  const float* eWhh1 = (const float*)d_in[7];
  const float* eb1   = (const float*)d_in[8];
  const float* dWih0 = (const float*)d_in[9];
  const float* dWhh0 = (const float*)d_in[10];
  const float* db0   = (const float*)d_in[11];
  const float* dWih1 = (const float*)d_in[12];
  const float* dWhh1 = (const float*)d_in[13];
  const float* db1   = (const float*)d_in[14];
  const float* outW  = (const float*)d_in[15];
  const float* outb  = (const float*)d_in[16];

  char* ws = (char*)d_ws;
  unsigned* bar = (unsigned*)ws;          // 256B header: barrier cnt/gen
  __bf16* w = (__bf16*)(ws + 256);
  size_t o = 0;
  __bf16* pEih0 = w + o; o += (size_t)G4 * D_;
  __bf16* pEhh0 = w + o; o += (size_t)G4 * H_;
  __bf16* pEih1 = w + o; o += (size_t)G4 * H_;
  __bf16* pEhh1 = w + o; o += (size_t)G4 * H_;
  __bf16* pDih0 = w + o; o += (size_t)G4 * D_;
  __bf16* pDhh0 = w + o; o += (size_t)G4 * H_;
  __bf16* pDih1 = w + o; o += (size_t)G4 * H_;
  __bf16* pDhh1 = w + o; o += (size_t)G4 * H_;
  __bf16* pOut  = w + o; o += (size_t)D_ * H_;
  o = (o + 63) & ~(size_t)63;
  __bf16* h0g  = w + o; o += (size_t)2 * B_ * H_;   // double-buffered
  __bf16* h1g  = w + o; o += (size_t)2 * B_ * H_;
  __bf16* din  = w + o; o += (size_t)B_ * D_;

  float* outp   = (float*)d_out;
  float* encout = outp + (size_t)B_ * T_ * D_;

  cvt_kernel<<<64, 256, 0, stream>>>(eWih0, pEih0, G4 * D_);
  cvt_kernel<<<64, 256, 0, stream>>>(eWhh0, pEhh0, G4 * H_);
  cvt_kernel<<<64, 256, 0, stream>>>(eWih1, pEih1, G4 * H_);
  cvt_kernel<<<64, 256, 0, stream>>>(eWhh1, pEhh1, G4 * H_);
  cvt_kernel<<<64, 256, 0, stream>>>(dWih0, pDih0, G4 * D_);
  cvt_kernel<<<64, 256, 0, stream>>>(dWhh0, pDhh0, G4 * H_);
  cvt_kernel<<<64, 256, 0, stream>>>(dWih1, pDih1, G4 * H_);
  cvt_kernel<<<64, 256, 0, stream>>>(dWhh1, pDhh1, G4 * H_);
  cvt_kernel<<<16, 256, 0, stream>>>(outW, pOut, D_ * H_);
  init_kernel<<<(B_ * H_ + 255) / 256, 256, 0, stream>>>(h0g, h1g, din, x, bar);

  lstm_persistent<<<NWG, TPB, 0, stream>>>(
      x, lengths, tf, eb0, eb1, db0, db1, outb,
      pEih0, pEhh0, pEih1, pEhh1, pDih0, pDhh0, pDih1, pDhh1, pOut,
      h0g, h1g, din, bar, bar + 1, outp, encout);
}
